// BiLSTM_3633542332540
// MI455X (gfx1250) — matrix operand order repaired and checked by reading, not compiled, not run on hardware
//
#include <hip/hip_runtime.h>
#include <hip/hip_bf16.h>

// ---------------- Problem constants ----------------
#define TT   128
#define BB   256
#define INDIM 512
#define HH   512
#define H4   2048   // 4*H

typedef __bf16 bf16_t;
typedef __attribute__((ext_vector_type(16))) __bf16 v16bf;
typedef __attribute__((ext_vector_type(8)))  __bf16 v8bf;
typedef __attribute__((ext_vector_type(8)))  float  v8f;

union V16 { v16bf v; v8bf h[2]; };

__device__ __forceinline__ bf16_t f2bf(float f) {
  union { float f; unsigned u; } in; in.f = f;
  unsigned u = in.u;
  unsigned r = u + 0x7FFFu + ((u >> 16) & 1u);   // round-to-nearest-even
  unsigned short s = (unsigned short)(r >> 16);
  union { unsigned short s; bf16_t b; } out; out.s = s;
  return out.b;
}

__device__ __forceinline__ float sigmoidf_(float x) {
  return 1.0f / (1.0f + __expf(-x));
}
__device__ __forceinline__ float tanhf_(float x) {
  x = fminf(fmaxf(x, -15.0f), 15.0f);
  float e = __expf(2.0f * x);
  return (e - 1.0f) / (e + 1.0f);
}

// ---------------- fp32 -> bf16 convert ----------------
__global__ void f32_to_bf16_kernel(const float* __restrict__ in,
                                   bf16_t* __restrict__ out, int n) {
  int i = (blockIdx.x * blockDim.x + threadIdx.x) * 4;
  if (i + 3 < n) {
    float4 v = *reinterpret_cast<const float4*>(in + i);
    out[i+0] = f2bf(v.x); out[i+1] = f2bf(v.y);
    out[i+2] = f2bf(v.z); out[i+3] = f2bf(v.w);
  } else {
    for (; i < n; ++i) out[i] = f2bf(in[i]);
  }
}

// Build combined bf16 weights: Wc[n][k] = k<in_dim ? Wih[n][k] : Whh[n][k-in_dim]
// Row n of Wc is the B-operand column n (gates = [x,h] @ Wc^T), so WMMA B-fragment
// loads are contiguous per lane.
__global__ void build_wc_kernel(const float* __restrict__ Wih,
                                const float* __restrict__ Whh,
                                bf16_t* __restrict__ Wc, int in_dim, int K) {
  int i = blockIdx.x * blockDim.x + threadIdx.x;
  int total = H4 * K;
  if (i >= total) return;
  int n = i / K, k = i % K;
  float v = (k < in_dim) ? Wih[n * in_dim + k] : Whh[n * HH + (k - in_dim)];
  Wc[i] = f2bf(v);
}

// ---------------- Fused BiLSTM step ----------------
// One kernel per timestep handles BOTH directions.
// Wave layout: 512 waves total: dir(2) x batch-supertile(8, 32 rows) x col-tile(32, 16 cols).
// Each wave owns a 32x16 (batch x hidden) patch and computes all 4 gates for it
// (8 f32 WMMA accumulators), then does the LSTM cell update in-register.
// h is double-buffered bf16 (no grid sync needed); c is f32, uniquely owned.
template <int IN_DIM>
__global__ __launch_bounds__(256) void lstm_step_kernel(
    const bf16_t* __restrict__ xin,     // [T][B][IN_DIM] bf16 layer input
    int t,                              // step 0..T-1
    const bf16_t* __restrict__ WcF,     // [4H][K] bf16, forward
    const bf16_t* __restrict__ WcB,     // backward
    const float*  __restrict__ biasF,   // [4H]
    const float*  __restrict__ biasB,
    const int*    __restrict__ lengths, // [B]
    float*  __restrict__ cState,        // [2][B][H] f32
    bf16_t* __restrict__ hBufs,         // [2 dir][2 parity][B][H] bf16
    bf16_t* __restrict__ outBf,         // [T][B][2H] bf16 (intermediate layers)
    float*  __restrict__ outF32)        // [T][B][2H] f32  (final layer)
{
  constexpr int K = IN_DIM + HH;
  const int lane  = threadIdx.x & 31;
  const int laneL = lane & 15;
  const int hi    = lane >> 4;
  const int wid   = blockIdx.x * (blockDim.x >> 5) + (threadIdx.x >> 5);
  const int dir   = wid >> 8;          // 0 = fwd, 1 = bwd
  const int rest  = wid & 255;
  const int mm    = rest >> 5;         // batch supertile (rows mm*32..+31)
  const int nt    = rest & 31;         // hidden col tile
  const int colg  = nt * 16 + laneL;   // hidden column 0..511

  const int t_eff = dir ? (TT - 1 - t) : t;
  const bf16_t* Wc   = dir ? WcB   : WcF;
  const float*  bias = dir ? biasB : biasF;

  float*        cPtr = cState + (size_t)dir * (BB * HH);
  const bf16_t* hRd  = hBufs + ((size_t)(dir * 2 + (t & 1)))     * (BB * HH);
  bf16_t*       hWr  = hBufs + ((size_t)(dir * 2 + ((t + 1) & 1))) * (BB * HH);
  const bf16_t* xt   = xin + (size_t)t_eff * BB * IN_DIM;

  const int r0 = mm * 32 + laneL;      // A rows for the two 16-row M blocks
  const int r1 = r0 + 16;

  // Accumulators init = bias broadcast down columns
  v8f acc[2][4];
  #pragma unroll
  for (int q = 0; q < 4; ++q) {
    float bv = bias[q * HH + colg];
    v8f z;
    #pragma unroll
    for (int j = 0; j < 8; ++j) z[j] = bv;
    acc[0][q] = z; acc[1][q] = z;
  }

  const bf16_t* wrow[4];
  #pragma unroll
  for (int q = 0; q < 4; ++q)
    wrow[q] = Wc + (size_t)(q * HH + colg) * K;

  for (int k0 = 0; k0 < K; k0 += 32) {
    // ----- A fragments (16x32 bf16, per ISA layout) -----
    const bf16_t* aBase; int kl, stride;
    if (k0 < IN_DIM) { aBase = xt;  kl = k0;          stride = IN_DIM; }
    else             { aBase = hRd; kl = k0 - IN_DIM; stride = HH;     }
    V16 a0, a1;
    {
      const bf16_t* p0 = aBase + (size_t)r0 * stride + kl + hi * 8;
      a0.h[0] = *reinterpret_cast<const v8bf*>(p0);        // K = k0+hi*8   .. +7
      a0.h[1] = *reinterpret_cast<const v8bf*>(p0 + 16);   // K = k0+16+hi*8.. +7
      const bf16_t* p1 = aBase + (size_t)r1 * stride + kl + hi * 8;
      a1.h[0] = *reinterpret_cast<const v8bf*>(p1);
      a1.h[1] = *reinterpret_cast<const v8bf*>(p1 + 16);
    }
    // ----- 4 gate B fragments + 8 WMMAs -----
    #pragma unroll
    for (int q = 0; q < 4; ++q) {
      V16 bf;
      const bf16_t* pw = wrow[q] + k0 + hi * 8;            // K = k0+hi*8..+7, k0+16+hi*8..+7
      bf.h[0] = *reinterpret_cast<const v8bf*>(pw);
      bf.h[1] = *reinterpret_cast<const v8bf*>(pw + 16);
      acc[0][q] = __builtin_amdgcn_wmma_f32_16x16x32_bf16(
          false, a0.v, false, bf.v, (short)0, acc[0][q], false, false);
      acc[1][q] = __builtin_amdgcn_wmma_f32_16x16x32_bf16(
          false, a1.v, false, bf.v, (short)0, acc[1][q], false, false);
    }
  }

  // ----- In-register LSTM cell update + masked writes -----
  #pragma unroll
  for (int mb = 0; mb < 2; ++mb) {
    const int rowBase = mm * 32 + mb * 16;
    #pragma unroll
    for (int j = 0; j < 8; ++j) {
      const int bg = rowBase + hi * 8 + j;       // C/D row: VGPR j -> M = j + 8*hi
      const bool active = t_eff < lengths[bg];
      const size_t sIdx = (size_t)bg * HH + colg;
      float hn = 0.0f;
      if (active) {
        float ig = sigmoidf_(acc[mb][0][j]);
        float fg = sigmoidf_(acc[mb][1][j]);
        float gg = tanhf_  (acc[mb][2][j]);
        float og = sigmoidf_(acc[mb][3][j]);
        float cn = fg * cPtr[sIdx] + ig * gg;
        cPtr[sIdx] = cn;
        hn = og * tanhf_(cn);
        hWr[sIdx] = f2bf(hn);
      } else {
        hWr[sIdx] = hRd[sIdx];                   // state frozen past seq end
      }
      const size_t oIdx = ((size_t)t_eff * BB + bg) * (2 * HH) + dir * HH + colg;
      if (outF32) outF32[oIdx] = hn;             // padded slots -> 0
      else        outBf[oIdx]  = f2bf(hn);
    }
  }
}

// ---------------- Host side ----------------
extern "C" void kernel_launch(void* const* d_in, const int* in_sizes, int n_in,
                              void* d_out, int out_size, void* d_ws, size_t ws_size,
                              hipStream_t stream) {
  (void)in_sizes; (void)n_in; (void)out_size; (void)ws_size;

  const float* x       = (const float*)d_in[0];
  const int*   lengths = (const int*)  d_in[1];
  // params flatten (insertion order): per layer: f{Wih,Whh,b}, b{Wih,Whh,b}
  const float* Wih[3][2]; const float* Whh[3][2]; const float* bia[3][2];
  for (int l = 0; l < 3; ++l)
    for (int d = 0; d < 2; ++d) {
      int base = 2 + l * 6 + d * 3;
      Wih[l][d] = (const float*)d_in[base + 0];
      Whh[l][d] = (const float*)d_in[base + 1];
      bia[l][d] = (const float*)d_in[base + 2];
    }

  // Workspace carve-up (256B aligned)
  char* ws = (char*)d_ws; size_t off = 0;
  auto carve = [&](size_t bytes) -> void* {
    void* p = ws + off;
    off += (bytes + 255) & ~(size_t)255;
    return p;
  };
  bf16_t* xbf = (bf16_t*)carve((size_t)TT * BB * INDIM * 2);     // 32 MiB
  bf16_t* y0  = (bf16_t*)carve((size_t)TT * BB * 2 * HH * 2);    // 64 MiB
  bf16_t* y1  = (bf16_t*)carve((size_t)TT * BB * 2 * HH * 2);    // 64 MiB
  bf16_t* Wc[3][2];
  const int Kl[3] = {INDIM + HH, 2 * HH + HH, 2 * HH + HH};      // 1024,1536,1536
  for (int l = 0; l < 3; ++l)
    for (int d = 0; d < 2; ++d)
      Wc[l][d] = (bf16_t*)carve((size_t)H4 * Kl[l] * 2);
  float*  cState = (float*) carve((size_t)2 * BB * HH * 4);      // [2][B][H]
  bf16_t* hBufs  = (bf16_t*)carve((size_t)4 * BB * HH * 2);      // [2][2][B][H]

  // 1) Convert input x to bf16
  {
    int n = TT * BB * INDIM;
    int blocks = (n / 4 + 255) / 256;
    f32_to_bf16_kernel<<<blocks, 256, 0, stream>>>(x, xbf, n);
  }
  // 2) Pack combined bf16 weights [Wih | Whh] per (layer, dir)
  for (int l = 0; l < 3; ++l) {
    int in_dim = (l == 0) ? INDIM : 2 * HH;
    int total = H4 * Kl[l];
    int blocks = (total + 255) / 256;
    for (int d = 0; d < 2; ++d)
      build_wc_kernel<<<blocks, 256, 0, stream>>>(Wih[l][d], Whh[l][d],
                                                  Wc[l][d], in_dim, Kl[l]);
  }

  // 3) Layers: zero states, then 128 fused bidirectional steps each
  const size_t stateBytes = (size_t)2 * BB * HH * 4 + (size_t)4 * BB * HH * 2;
  const dim3 grid(64), block(256);  // 512 waves = 2 dir x 8 Mtiles x 32 Ntiles

  for (int l = 0; l < 3; ++l) {
    hipMemsetAsync(cState, 0, stateBytes, stream);  // c + both h parities
    const bf16_t* xin = (l == 0) ? xbf : (l == 1 ? y0 : y1);
    bf16_t* outBf = (l == 0) ? y0 : (l == 1 ? y1 : nullptr);
    float*  outF  = (l == 2) ? (float*)d_out : nullptr;
    for (int t = 0; t < TT; ++t) {
      if (l == 0)
        lstm_step_kernel<INDIM><<<grid, block, 0, stream>>>(
            xin, t, Wc[l][0], Wc[l][1], bia[l][0], bia[l][1],
            lengths, cState, hBufs, outBf, outF);
      else
        lstm_step_kernel<2 * HH><<<grid, block, 0, stream>>>(
            xin, t, Wc[l][0], Wc[l][1], bia[l][0], bia[l][1],
            lengths, cState, hBufs, outBf, outF);
    }
  }
}
